// AttentionBlock_2121713844615
// MI455X (gfx1250) — matrix operand order repaired and checked by reading, not compiled, not run on hardware
//
#include <hip/hip_runtime.h>

#define C_CH 128
#define HW   65536
#define BN   128
#define LDP  136    // conv LDS leading dim (128 + 8 pad, halfs)
#define LDA  264    // attn f16 LDS leading dim (256 + 8 pad, halfs)
#define LDS_S 264   // attn f32 score LDS leading dim (floats)

typedef __attribute__((ext_vector_type(16))) _Float16 v16h;
typedef __attribute__((ext_vector_type(8)))  float    v8f;
typedef int v4i __attribute__((vector_size(16)));

// ---- CDNA5 async global->LDS staging (ASYNCcnt path), with plain fallback --
#if defined(__HIP_DEVICE_COMPILE__) && __has_builtin(__builtin_amdgcn_global_load_async_to_lds_b128)
#define HAVE_ASYNC_LDS 1
#else
#define HAVE_ASYNC_LDS 0
#endif

__device__ __forceinline__ void async_copy16(void* lds_dst, const void* gsrc) {
#if HAVE_ASYNC_LDS
    __builtin_amdgcn_global_load_async_to_lds_b128(
        (__attribute__((address_space(1))) v4i*)gsrc,
        (__attribute__((address_space(3))) v4i*)lds_dst, 0, 0);
#else
    *(uint4*)lds_dst = *(const uint4*)gsrc;
#endif
}

__device__ __forceinline__ void async_join() {
#if HAVE_ASYNC_LDS
#if __has_builtin(__builtin_amdgcn_s_wait_asynccnt)
    __builtin_amdgcn_s_wait_asynccnt(0);
#else
    asm volatile("s_wait_asynccnt 0x0" ::: "memory");
#endif
#endif
}

// ---- WMMA helpers ----------------------------------------------------------
__device__ __forceinline__ v8f wmma_f16(v16h a, v16h b, v8f c) {
    return __builtin_amdgcn_wmma_f32_16x16x32_f16(
        false, a, false, b, (short)0, c, false, false);
}

// A fragment: 16x32 f16 tile, row-major source, leading dim ld.
// lane&15 = M row; lane>>4 = K half (k = 16*half + e). Contiguous 32B per lane.
__device__ __forceinline__ v16h frag_a(const _Float16* src, int ld) {
    const int lane = threadIdx.x & 31;
    const _Float16* r = src + (lane & 15) * ld + ((lane >> 4) & 1) * 16;
    v16h a;
#pragma unroll
    for (int e = 0; e < 8; ++e) { a[e] = r[e]; a[e + 8] = r[8 + e]; }
    return a;
}

// B fragment: 32x16, B[k][n] row-major (strided per-lane reads).
__device__ __forceinline__ v16h frag_b(const _Float16* src, int ld) {
    const int lane = threadIdx.x & 31;
    const _Float16* c0 = src + (((lane >> 4) & 1) * 16) * ld + (lane & 15);
    v16h b;
#pragma unroll
    for (int e = 0; e < 16; ++e) b[e] = c0[e * ld];
    return b;
}

// B fragment with B = M^T (B[k][n] = M[n][k]); contiguous 32B per lane.
__device__ __forceinline__ v16h frag_bt(const _Float16* srcM, int ld) {
    const int lane = threadIdx.x & 31;
    const _Float16* r = srcM + (lane & 15) * ld + ((lane >> 4) & 1) * 16;
    v16h b;
#pragma unroll
    for (int e = 0; e < 16; ++e) b[e] = r[e];
    return b;
}

// ---------------------------------------------------------------------------
// 1x1 conv as GEMM: out[b][o][p] = sum_c W[o][c]*X[b][c][p] + bias[o]
// 8 waves; wave = 16-pixel N-slice, loops all 8 M-tiles (B-frag loaded once
// per K-chunk per wave; replicated A-frags are cheap contiguous b128 loads).
// sel==2 (v projection) stores transposed per-(b,c) image: vT[w][h].
// ---------------------------------------------------------------------------
template <bool IN_HALF, bool OUT_HALF>
__global__ __launch_bounds__(256) void conv1x1_gemm(
    const float* __restrict__ xf, const _Float16* __restrict__ xh,
    const float* __restrict__ W0, const float* __restrict__ W1,
    const float* __restrict__ W2, const float* __restrict__ B0,
    const float* __restrict__ B1, const float* __restrict__ B2,
    _Float16* __restrict__ o0, _Float16* __restrict__ o1,
    _Float16* __restrict__ o2, float* __restrict__ of)
{
    __shared__ _Float16 sW[C_CH * LDP];
    __shared__ _Float16 sX[C_CH * LDP];

    const int t   = threadIdx.x;
    const int b   = blockIdx.y;
    const int p0  = blockIdx.x * BN;
    const int sel = blockIdx.z;

    const float* W  = (sel == 0) ? W0 : (sel == 1 ? W1 : W2);
    const float* Bi = (sel == 0) ? B0 : (sel == 1 ? B1 : B2);
    _Float16*    oh = (sel == 0) ? o0 : (sel == 1 ? o1 : o2);

    // Stage weight 128x128 f32 -> f16 (coalesced, 64/thread).
    {
        const int i0 = t * 64;
#pragma unroll 8
        for (int i = 0; i < 64; ++i) {
            const int idx = i0 + i;
            sW[(idx >> 7) * LDP + (idx & 127)] = (_Float16)W[idx];
        }
    }
    // Stage X tile (128 channels x 128 pixels).
    {
        const int c  = t >> 1;
        const int ps = (t & 1) * 64;
        const size_t gbase = ((size_t)(b * C_CH + c)) * HW + p0 + ps;
        if (IN_HALF) {
            const _Float16* src = xh + gbase;
            _Float16* dst = &sX[c * LDP + ps];
#pragma unroll
            for (int i = 0; i < 8; ++i) async_copy16(dst + i * 8, src + i * 8);
            async_join();
        } else {
            const float* src = xf + gbase;
#pragma unroll 8
            for (int i = 0; i < 64; ++i) sX[c * LDP + ps + i] = (_Float16)src[i];
        }
    }
    __syncthreads();

    const int wv = t >> 5, lane = t & 31;
    const int n0 = wv * 16;                 // this wave's 16-pixel column slice

    v8f acc[8] = {};
#pragma unroll
    for (int kc = 0; kc < 4; ++kc) {        // K = 128 in 4 chunks of 32
        const v16h bf = frag_b(&sX[(kc * 32) * LDP + n0], LDP);
#pragma unroll
        for (int mt = 0; mt < 8; ++mt) {
            const v16h a = frag_a(&sW[(mt * 16) * LDP + kc * 32], LDP);
            acc[mt] = wmma_f16(a, bf, acc[mt]);
        }
    }

    const int nn = lane & 15;
    const int mh = (lane >> 4) & 1;
    const int p  = p0 + n0 + nn;            // pixel this lane writes
#pragma unroll
    for (int mt = 0; mt < 8; ++mt) {
#pragma unroll
        for (int j = 0; j < 8; ++j) {
            const int m = mt * 16 + mh * 8 + j;
            const float val = acc[mt][j] + Bi[m];
            if (OUT_HALF) {
                size_t idx;
                if (sel == 2) {             // v: store transposed image w*H+h
                    const int h = p >> 8, wimg = p & 255;
                    idx = ((size_t)(b * C_CH + m)) * HW + (size_t)wimg * 256 + h;
                } else {
                    idx = ((size_t)(b * C_CH + m)) * HW + p;
                }
                oh[idx] = (_Float16)val;
            } else {
                of[((size_t)(b * C_CH + m)) * HW + p] = val;
            }
        }
    }
}

// ---------------------------------------------------------------------------
// Per-(b,c) row attention. Block = 256 thr (8 waves) handles 64 rows.
// Phase1: S = q @ k^T. Softmax rows. Phase2: out = P @ v (v staged as v^T so
// both phases use contiguous frag_bt reads). ~231 KB LDS -> 1 WG / WGP.
// blockIdx: x = 64-row tile (0..3), y = b*C + c.
// ---------------------------------------------------------------------------
__global__ __launch_bounds__(256) void attn_kernel(
    const _Float16* __restrict__ qb, const _Float16* __restrict__ kb,
    const _Float16* __restrict__ vtb, _Float16* __restrict__ ab)
{
    __shared__ float    sS [64 * LDS_S];    // scores f32
    __shared__ _Float16 sKV[256 * LDA];     // k, then v^T
    __shared__ _Float16 sQP[64 * LDA];      // q rows, then P f16

    const int t  = threadIdx.x;
    const int bc = blockIdx.y;
    const int r0 = blockIdx.x * 64;
    const size_t base = (size_t)bc * HW;

    // Async-stage q rows [r0, r0+64): 2048 16B chunks, 8/thread.
    {
        const _Float16* src = qb + base + (size_t)r0 * 256;
#pragma unroll
        for (int i = 0; i < 8; ++i) {
            const int ch = t + i * 256, row = ch >> 5, u = ch & 31;
            async_copy16(&sQP[row * LDA + u * 8], src + row * 256 + u * 8);
        }
    }
    // Async-stage full k (256x256): 8192 chunks, 32/thread.
    {
        const _Float16* src = kb + base;
#pragma unroll
        for (int i = 0; i < 32; ++i) {
            const int ch = t + i * 256, row = ch >> 5, u = ch & 31;
            async_copy16(&sKV[row * LDA + u * 8], src + row * 256 + u * 8);
        }
    }
    async_join();
    __syncthreads();

    const int wv = t >> 5, lane = t & 31;
    const int n0 = wv * 32;                 // wave owns two 16-col N-tiles
    const int nn = lane & 15, mh = (lane >> 4) & 1;

    // Phase 1: S = q @ k^T
    {
        v8f acc[8] = {};
#pragma unroll
        for (int kc = 0; kc < 8; ++kc) {
            const v16h b0 = frag_bt(&sKV[(n0)      * LDA + kc * 32], LDA);
            const v16h b1 = frag_bt(&sKV[(n0 + 16) * LDA + kc * 32], LDA);
#pragma unroll
            for (int mt = 0; mt < 4; ++mt) {
                const v16h a = frag_a(&sQP[(mt * 16) * LDA + kc * 32], LDA);
                acc[mt * 2]     = wmma_f16(a, b0, acc[mt * 2]);
                acc[mt * 2 + 1] = wmma_f16(a, b1, acc[mt * 2 + 1]);
            }
        }
#pragma unroll
        for (int mt = 0; mt < 4; ++mt)
#pragma unroll
            for (int ntl = 0; ntl < 2; ++ntl)
#pragma unroll
                for (int j = 0; j < 8; ++j)
                    sS[(mt * 16 + mh * 8 + j) * LDS_S + n0 + ntl * 16 + nn] =
                        acc[mt * 2 + ntl][j];
    }
    __syncthreads();

    // Kick off async v^T stage now; it overlaps the softmax below.
    {
        const _Float16* src = vtb + base;
#pragma unroll
        for (int i = 0; i < 32; ++i) {
            const int ch = t + i * 256, row = ch >> 5, u = ch & 31;
            async_copy16(&sKV[row * LDA + u * 8], src + row * 256 + u * 8);
        }
    }

    // Softmax over each 256-wide row; 4 threads/row, stride-4 interleave
    // (keeps a wave spread over 32 LDS banks), shfl_xor combine.
    {
        const int row = t >> 2, sub = t & 3;
        const float* r = &sS[row * LDS_S];
        float mx = -3.4e38f;
#pragma unroll 8
        for (int i = 0; i < 64; ++i) mx = fmaxf(mx, r[sub + 4 * i]);
        mx = fmaxf(mx, __shfl_xor(mx, 1, 32));
        mx = fmaxf(mx, __shfl_xor(mx, 2, 32));
        float sum = 0.f;
#pragma unroll 8
        for (int i = 0; i < 64; ++i) sum += __expf(r[sub + 4 * i] - mx);
        sum += __shfl_xor(sum, 1, 32);
        sum += __shfl_xor(sum, 2, 32);
        const float inv = 1.0f / sum;
        _Float16* pr = &sQP[row * LDA];     // q is dead -> P lives here
#pragma unroll 8
        for (int i = 0; i < 64; ++i) {
            const int col = sub + 4 * i;
            pr[col] = (_Float16)(__expf(r[col] - mx) * inv);
        }
    }
    async_join();
    __syncthreads();

    // Phase 2: out = P @ v, with v^T in LDS -> contiguous frag_bt reads.
    {
        v8f acc[8] = {};
#pragma unroll
        for (int kc = 0; kc < 8; ++kc) {
            const v16h b0 = frag_bt(&sKV[(n0)      * LDA + kc * 32], LDA);
            const v16h b1 = frag_bt(&sKV[(n0 + 16) * LDA + kc * 32], LDA);
#pragma unroll
            for (int mt = 0; mt < 4; ++mt) {
                const v16h a = frag_a(&sQP[(mt * 16) * LDA + kc * 32], LDA);
                acc[mt * 2]     = wmma_f16(a, b0, acc[mt * 2]);
                acc[mt * 2 + 1] = wmma_f16(a, b1, acc[mt * 2 + 1]);
            }
        }
#pragma unroll
        for (int mt = 0; mt < 4; ++mt)
#pragma unroll
            for (int ntl = 0; ntl < 2; ++ntl)
#pragma unroll
                for (int j = 0; j < 8; ++j) {
                    const int row = r0 + mt * 16 + mh * 8 + j;
                    ab[base + (size_t)row * 256 + n0 + ntl * 16 + nn] =
                        (_Float16)acc[mt * 2 + ntl][j];
                }
    }
}

// ---------------------------------------------------------------------------
extern "C" void kernel_launch(void* const* d_in, const int* in_sizes, int n_in,
                              void* d_out, int out_size, void* d_ws, size_t ws_size,
                              hipStream_t stream) {
    const float* x  = (const float*)d_in[0];
    const float* wq = (const float*)d_in[1];
    const float* bq = (const float*)d_in[2];
    const float* wk = (const float*)d_in[3];
    const float* bk = (const float*)d_in[4];
    const float* wv = (const float*)d_in[5];
    const float* bv = (const float*)d_in[6];
    const float* wo = (const float*)d_in[7];
    const float* bo = (const float*)d_in[8];
    float* out = (float*)d_out;

    const size_t per = (size_t)4 * C_CH * HW;   // elements per f16 tensor
    _Float16* qb  = (_Float16*)d_ws;
    _Float16* kb  = qb + per;
    _Float16* vtb = kb + per;                    // v stored transposed
    _Float16* ab  = vtb + per;

    // 1) Q,K,V projections (f32 x -> f16 ws; v written transposed), bias fused.
    dim3 g1(HW / BN, 4, 3);
    conv1x1_gemm<false, true><<<g1, 256, 0, stream>>>(
        x, nullptr, wq, wk, wv, bq, bk, bv, qb, kb, vtb, nullptr);

    // 2) Per-(b,c) row attention.
    dim3 g2(256 / 64, 4 * C_CH);
    attn_kernel<<<g2, 256, 0, stream>>>(qb, kb, vtb, ab);

    // 3) Output projection (f16 ws -> f32 out), bias fused.
    dim3 g3(HW / BN, 4, 1);
    conv1x1_gemm<true, false><<<g3, 256, 0, stream>>>(
        nullptr, ab, wo, wo, wo, bo, bo, bo, nullptr, nullptr, nullptr, out);
}
